// LSTMAcceptor_68496138437207
// MI455X (gfx1250) — compile-verified, matrix operand order checked by reading, not run on hardware
//
#include <hip/hip_runtime.h>
#include <cstddef>
#include <cstdint>

#define BATCH 64
#define SEQ   512
#define EMBD  512
#define HIDD  1024
#define KTOT  1536   /* EMBD + HIDD */
#define NBLK  64     /* blocks in the persistent recurrence grid */

typedef __attribute__((ext_vector_type(16))) __bf16        v16bf;
typedef __attribute__((ext_vector_type(8)))  float         v8f;
typedef __attribute__((ext_vector_type(4)))  float         f32x4;
typedef __attribute__((ext_vector_type(4)))  unsigned int  u32x4;

union Frag {
  v16bf          v;
  unsigned short us[16];
  u32x4          q[2];
};

static __device__ __forceinline__ unsigned short f2bf(float f) {
  unsigned int u = __float_as_uint(f);
  u += 0x7FFFu + ((u >> 16) & 1u);            // round-to-nearest-even
  return (unsigned short)(u >> 16);
}
static __device__ __forceinline__ float sigm(float x) {
  return 1.0f / (1.0f + __expf(-x));
}

// ---------------------------------------------------------------------------
// Prep A: Wc[r][0:512]=W_ih[r], Wc[r][512:1536]=W_hh[r] (bf16);
//         bsum = b_ih + b_hh; zero h0/h1 (bf16), c (f32), barrier counter.
// ---------------------------------------------------------------------------
__global__ __launch_bounds__(256) void prep_kernel(
    const float* __restrict__ W_ih, const float* __restrict__ W_hh,
    const float* __restrict__ b_ih, const float* __restrict__ b_hh,
    unsigned short* __restrict__ Wc, float* __restrict__ bsum,
    unsigned short* __restrict__ h0, unsigned short* __restrict__ h1,
    float* __restrict__ cst, unsigned* __restrict__ gen)
{
  const int e = blockIdx.x * 256 + threadIdx.x;
  const int total = 4 * HIDD * KTOT;
  if (e < total) {
    const int r = e / KTOT;
    const int k = e - r * KTOT;
    const float v = (k < EMBD) ? W_ih[(size_t)r * EMBD + k]
                               : W_hh[(size_t)r * HIDD + (k - EMBD)];
    Wc[e] = f2bf(v);
  }
  if (e < 4 * HIDD) bsum[e] = b_ih[e] + b_hh[e];
  if (e < BATCH * HIDD) { h0[e] = 0; h1[e] = 0; cst[e] = 0.0f; }
  if (e == 0) *gen = 0u;
}

// ---------------------------------------------------------------------------
// Prep B: gather + convert embeddings once, in parallel (off the serial path):
//   seq[b][t][0:512] = bf16(emb_table[x[b,t]])  (zeros for padding_idx 0)
// ---------------------------------------------------------------------------
__global__ __launch_bounds__(256) void gather_kernel(
    const int*   __restrict__ x,          // [BATCH, SEQ]
    const float* __restrict__ emb_table,  // [VOCAB, EMBD]
    unsigned short* __restrict__ seq)     // [BATCH, SEQ, EMBD] bf16
{
  const size_t tid = (size_t)blockIdx.x * 256 + threadIdx.x;
  const size_t e8  = tid * 8;
  if (e8 >= (size_t)BATCH * SEQ * EMBD) return;
  const size_t bt = e8 / EMBD;                 // (b*SEQ + t), matches x layout
  const int    k  = (int)(e8 - bt * EMBD);
  const int    idx = x[bt];
  const float  esc = (idx == 0) ? 0.0f : 1.0f; // padding_idx row -> zeros
  const float* src = emb_table + (size_t)idx * EMBD + k;
  f32x4 a = *(const f32x4*)(src);
  f32x4 b = *(const f32x4*)(src + 4);
  union { u32x4 q; unsigned short us[8]; } o;
#pragma unroll
  for (int i = 0; i < 4; ++i) {
    o.us[i]     = f2bf(a[i] * esc);
    o.us[4 + i] = f2bf(b[i] * esc);
  }
  *(u32x4*)(seq + e8) = o.q;
}

// ---------------------------------------------------------------------------
// Persistent LSTM: ALL 512 timesteps in ONE launch.
// Per step: gates = [seq_t | h_t] @ Wc^T + bsum (bf16 WMMA, f32 acc), then
// lane-local cell update, then an agent-scope release/acquire inter-block
// barrier on a monotonic generation counter (64 blocks, trivially co-resident:
// 256 waves total).
// Grid: 64 blocks (one 16-wide hidden-unit tile each), 128 threads (4 waves,
// one 16-row M tile each; a wave computes the SAME tile in all four gates, so
// i,f,g,o for any (row,unit) land in the same lane/element of 4 accumulators).
// ---------------------------------------------------------------------------
__global__ __launch_bounds__(128) void lstm_persistent_kernel(
    const unsigned short* __restrict__ seq,   // [BATCH, SEQ, EMBD] bf16
    const unsigned short* __restrict__ Wc,    // [4H, KTOT] bf16
    const float*          __restrict__ bsum,  // [4H]
    unsigned short*       __restrict__ h0,    // [BATCH, HIDD] bf16
    unsigned short*       __restrict__ h1,    // [BATCH, HIDD] bf16
    float*                __restrict__ cst,   // [BATCH, HIDD] f32
    unsigned*             __restrict__ gen)   // barrier generation counter
{
  const int lane = threadIdx.x & 31;
  const int wave = threadIdx.x >> 5;          // 0..3 -> M tile
  const int jt   = blockIdx.x;                // 0..63 -> hidden-unit tile
  const int nloc = lane & 15;
  const int kh   = lane >> 4;                 // K-half select (A/B layout)

  const int arow = (wave << 4) + nloc;        // batch row this lane feeds A
  const int ncol = (jt << 4) + nloc;          // hidden unit (column in gate)

  const unsigned short* wrow[4];
#pragma unroll
  for (int g = 0; g < 4; ++g)
    wrow[g] = Wc + (size_t)(g * HIDD + ncol) * KTOT;

  const float bi = bsum[0 * HIDD + ncol];
  const float bf = bsum[1 * HIDD + ncol];
  const float bg = bsum[2 * HIDD + ncol];
  const float bo = bsum[3 * HIDD + ncol];
  const int rbase = (wave << 4) + (kh << 3);  // C/D layout: M = v + 8*kh

  for (int t = 0; t < SEQ; ++t) {
    const unsigned short* hin  = (t & 1) ? h1 : h0;
    unsigned short*       hout = (t & 1) ? h0 : h1;
    const unsigned short* seqrow = seq + ((size_t)arow * SEQ + t) * EMBD;
    const unsigned short* hr     = hin + (size_t)arow * HIDD;

    // Warm L2 with next timestep's A row while we compute this one.
    __builtin_prefetch(seqrow + EMBD, 0, 0);

    const v8f vzero = {};
    v8f acc[4] = {vzero, vzero, vzero, vzero};

    // ---- K over the embedding half (pre-converted bf16) ----
#pragma unroll 2
    for (int k0 = 0; k0 < EMBD; k0 += 32) {
      const int kA0 = k0 + (kh << 3);
      const int kA1 = kA0 + 16;
      Frag a, b0, b1, b2, b3;
      a.q[0]  = *(const u32x4*)(seqrow + kA0);
      a.q[1]  = *(const u32x4*)(seqrow + kA1);
      b0.q[0] = *(const u32x4*)(wrow[0] + kA0);
      b0.q[1] = *(const u32x4*)(wrow[0] + kA1);
      b1.q[0] = *(const u32x4*)(wrow[1] + kA0);
      b1.q[1] = *(const u32x4*)(wrow[1] + kA1);
      b2.q[0] = *(const u32x4*)(wrow[2] + kA0);
      b2.q[1] = *(const u32x4*)(wrow[2] + kA1);
      b3.q[0] = *(const u32x4*)(wrow[3] + kA0);
      b3.q[1] = *(const u32x4*)(wrow[3] + kA1);
      acc[0] = __builtin_amdgcn_wmma_f32_16x16x32_bf16(false, a.v, false, b0.v,
                                                       (short)0, acc[0], false, false);
      acc[1] = __builtin_amdgcn_wmma_f32_16x16x32_bf16(false, a.v, false, b1.v,
                                                       (short)0, acc[1], false, false);
      acc[2] = __builtin_amdgcn_wmma_f32_16x16x32_bf16(false, a.v, false, b2.v,
                                                       (short)0, acc[2], false, false);
      acc[3] = __builtin_amdgcn_wmma_f32_16x16x32_bf16(false, a.v, false, b3.v,
                                                       (short)0, acc[3], false, false);
    }

    // ---- K over the recurrent half (bf16 h) ----
#pragma unroll 2
    for (int k0 = EMBD; k0 < KTOT; k0 += 32) {
      const int kA0 = k0 + (kh << 3);
      const int kA1 = kA0 + 16;
      Frag a, b0, b1, b2, b3;
      a.q[0]  = *(const u32x4*)(hr + (kA0 - EMBD));
      a.q[1]  = *(const u32x4*)(hr + (kA1 - EMBD));
      b0.q[0] = *(const u32x4*)(wrow[0] + kA0);
      b0.q[1] = *(const u32x4*)(wrow[0] + kA1);
      b1.q[0] = *(const u32x4*)(wrow[1] + kA0);
      b1.q[1] = *(const u32x4*)(wrow[1] + kA1);
      b2.q[0] = *(const u32x4*)(wrow[2] + kA0);
      b2.q[1] = *(const u32x4*)(wrow[2] + kA1);
      b3.q[0] = *(const u32x4*)(wrow[3] + kA0);
      b3.q[1] = *(const u32x4*)(wrow[3] + kA1);
      acc[0] = __builtin_amdgcn_wmma_f32_16x16x32_bf16(false, a.v, false, b0.v,
                                                       (short)0, acc[0], false, false);
      acc[1] = __builtin_amdgcn_wmma_f32_16x16x32_bf16(false, a.v, false, b1.v,
                                                       (short)0, acc[1], false, false);
      acc[2] = __builtin_amdgcn_wmma_f32_16x16x32_bf16(false, a.v, false, b2.v,
                                                       (short)0, acc[2], false, false);
      acc[3] = __builtin_amdgcn_wmma_f32_16x16x32_bf16(false, a.v, false, b3.v,
                                                       (short)0, acc[3], false, false);
    }

    // ---- Cell update: i,f,g,o for (row,col) in the same lane/element ----
#pragma unroll
    for (int v = 0; v < 8; ++v) {
      const size_t off = (size_t)(rbase + v) * HIDD + ncol;
      const float gi = acc[0][v] + bi;
      const float gf = acc[1][v] + bf;
      const float gg = acc[2][v] + bg;
      const float go = acc[3][v] + bo;
      const float cn = sigm(gf) * cst[off] + sigm(gi) * tanhf(gg);
      const float hn = sigm(go) * tanhf(cn);
      cst[off]  = cn;
      hout[off] = f2bf(hn);
    }

    // ---- Inter-block barrier: release h_{t+1}, acquire all blocks' h ----
    __threadfence();
    __syncthreads();
    if (threadIdx.x == 0) {
      __hip_atomic_fetch_add(gen, 1u, __ATOMIC_RELEASE, __HIP_MEMORY_SCOPE_AGENT);
      const unsigned need = (unsigned)(t + 1) * (unsigned)NBLK;
      while (__hip_atomic_load(gen, __ATOMIC_ACQUIRE, __HIP_MEMORY_SCOPE_AGENT) < need)
        __builtin_amdgcn_s_sleep(1);
    }
    __syncthreads();
  }
}

// ---------------------------------------------------------------------------
// Head MLP: hidden = relu(h @ W1^T + b1) ; out = hidden @ W2^T + b2
// ---------------------------------------------------------------------------
__global__ __launch_bounds__(256) void head1_kernel(
    const unsigned short* __restrict__ h, const float* __restrict__ W1,
    const float* __restrict__ b1, float* __restrict__ hidden)
{
  const int tid = blockIdx.x * 256 + threadIdx.x;   // 64*64 threads
  if (tid >= BATCH * 64) return;
  const int r = tid >> 6, j = tid & 63;
  const unsigned short* hr = h + (size_t)r * HIDD;
  const float* wr = W1 + (size_t)j * HIDD;
  float a = b1[j];
  for (int k = 0; k < HIDD; k += 8) {
    u32x4 hv = *(const u32x4*)(hr + k);
    f32x4 w0 = *(const f32x4*)(wr + k);
    f32x4 w1 = *(const f32x4*)(wr + k + 4);
#pragma unroll
    for (int i = 0; i < 4; ++i) {
      const float lo = __uint_as_float(hv[i] << 16);          // element 2i
      const float hi = __uint_as_float(hv[i] & 0xFFFF0000u);  // element 2i+1
      const float we = (i < 2) ? w0[2 * (i & 1)]     : w1[2 * (i & 1)];
      const float wo = (i < 2) ? w0[2 * (i & 1) + 1] : w1[2 * (i & 1) + 1];
      a += lo * we + hi * wo;
    }
  }
  hidden[tid] = fmaxf(a, 0.0f);
}

__global__ __launch_bounds__(64) void head2_kernel(
    const float* __restrict__ hidden, const float* __restrict__ W2,
    const float* __restrict__ b2, float* __restrict__ out)
{
  const int r = threadIdx.x;                        // 64 threads
  float a = b2[0];
  for (int j = 0; j < 64; ++j) a += hidden[r * 64 + j] * W2[j];
  out[r] = a;
}

// ---------------------------------------------------------------------------
extern "C" void kernel_launch(void* const* d_in, const int* in_sizes, int n_in,
                              void* d_out, int out_size, void* d_ws, size_t ws_size,
                              hipStream_t stream) {
  (void)in_sizes; (void)n_in; (void)out_size; (void)ws_size;

  const int*   x    = (const int*)  d_in[0];
  /* d_in[1] = lengths : unused by the reference */
  const float* emb  = (const float*)d_in[2];
  const float* W_ih = (const float*)d_in[3];
  const float* b_ih = (const float*)d_in[4];
  const float* W_hh = (const float*)d_in[5];
  const float* b_hh = (const float*)d_in[6];
  const float* W1   = (const float*)d_in[7];
  const float* b1   = (const float*)d_in[8];
  const float* W2   = (const float*)d_in[9];
  const float* b2   = (const float*)d_in[10];
  float* out = (float*)d_out;

  char* ws = (char*)d_ws;
  size_t off = 0;
  auto carve = [&](size_t bytes) -> char* {
    char* p = ws + off;
    off = (off + bytes + 255) & ~(size_t)255;
    return p;
  };
  unsigned short* Wc     = (unsigned short*)carve((size_t)4 * HIDD * KTOT * 2);    // 12.6 MB
  unsigned short* seqb   = (unsigned short*)carve((size_t)BATCH * SEQ * EMBD * 2); // 32 MB
  float*          bsum   = (float*)         carve((size_t)4 * HIDD * 4);
  unsigned short* h0     = (unsigned short*)carve((size_t)BATCH * HIDD * 2);
  unsigned short* h1     = (unsigned short*)carve((size_t)BATCH * HIDD * 2);
  float*          cst    = (float*)         carve((size_t)BATCH * HIDD * 4);
  float*          hidden = (float*)         carve((size_t)BATCH * 64 * 4);
  unsigned*       gen    = (unsigned*)      carve(256);

  // 1) weight conversion + state/barrier init (parallel)
  {
    const int total = 4 * HIDD * KTOT;
    prep_kernel<<<(total + 255) / 256, 256, 0, stream>>>(
        W_ih, W_hh, b_ih, b_hh, Wc, bsum, h0, h1, cst, gen);
  }
  // 2) embedding gather + bf16 convert for ALL timesteps (parallel)
  {
    const size_t nthreads = (size_t)BATCH * SEQ * EMBD / 8;   // 2,097,152
    gather_kernel<<<(unsigned)((nthreads + 255) / 256), 256, 0, stream>>>(
        x, emb, seqb);
  }

  // 3) entire 512-step recurrence in ONE persistent launch
  //    (inter-block barrier per step; final h lands in h0 since SEQ is even)
  lstm_persistent_kernel<<<NBLK, 128, 0, stream>>>(seqb, Wc, bsum, h0, h1, cst, gen);

  // 4) head MLP
  head1_kernel<<<(BATCH * 64 + 255) / 256, 256, 0, stream>>>(h0, W1, b1, hidden);
  head2_kernel<<<1, 64, 0, stream>>>(hidden, W2, b2, out);
}
